// InfoAggregation_20839181320891
// MI455X (gfx1250) — matrix operand order repaired and checked by reading, not compiled, not run on hardware
//
#include <hip/hip_runtime.h>
#include <hip/hip_bf16.h>

typedef __bf16 bf16_t;
typedef __attribute__((ext_vector_type(16))) __bf16 v16bf;
typedef __attribute__((ext_vector_type(8)))  __bf16 v8bf;
typedef __attribute__((ext_vector_type(4)))  __bf16 v4bf;
typedef __attribute__((ext_vector_type(8)))  float  v8f;

#define C_DIM 256
#define N_TOK 2304
#define SZ1   (C_DIM * N_TOK)   // 589824 elems: one [c,N] tensor for one batch
#define TSZ   (8 * SZ1)         // 4718592 elems: one projected tensor, all batches

__device__ __forceinline__ v8f wmma_bf16(v16bf a, v16bf b, v8f c) {
  // D = A(16x32 bf16) * B(32x16 bf16) + C(16x16 f32)
  return __builtin_amdgcn_wmma_f32_16x16x32_bf16(false, a, false, b, (short)0, c,
                                                 false, false);
}

// ---------------------------------------------------------------------------
// Kernel 1: projections.  Yt = W @ X per (batch, proj).
//   proj 0: Q = Wq @ Xq   -> stored [n][d]  (row-major per token)
//   proj 1: K1 = Wk @ Xf  -> stored [d][n]
//   proj 2: K2 = Wk @ Xm  -> stored [d][n]
//   proj 3: V = Wv @ Xf   -> stored [n][d]
// grid (9 ntiles, 4 dtiles, 32 = batch*4+proj), 128 threads (4 waves).
// Each wave: 16 d-rows x 256 n-cols, K-loop over 256 in chunks of 32.
// ---------------------------------------------------------------------------
__global__ __launch_bounds__(128) void proj_kernel(
    const float* __restrict__ qry, const float* __restrict__ mask,
    const float* __restrict__ fts,
    const float* __restrict__ Wq, const float* __restrict__ Wk,
    const float* __restrict__ Wv, bf16_t* __restrict__ wsb)
{
  const int proj  = blockIdx.z & 3;
  const int batch = blockIdx.z >> 2;
  const float* X; const float* W; bf16_t* Y; bool rowmajor;
  if (proj == 0)      { X = qry;  W = Wq; Y = wsb + (size_t)0 * TSZ; rowmajor = true;  }
  else if (proj == 1) { X = fts;  W = Wk; Y = wsb + (size_t)1 * TSZ; rowmajor = false; }
  else if (proj == 2) { X = mask; W = Wk; Y = wsb + (size_t)2 * TSZ; rowmajor = false; }
  else                { X = fts;  W = Wv; Y = wsb + (size_t)3 * TSZ; rowmajor = true;  }
  X += (size_t)batch * SZ1;
  Y += (size_t)batch * SZ1;

  const int lane = threadIdx.x & 31;
  const int wv   = threadIdx.x >> 5;
  const int lo   = lane & 15;
  const int hi   = lane >> 4;
  const int d0   = blockIdx.y * 64 + wv * 16;
  const int n0   = blockIdx.x * 256;

  v8f acc[16];
#pragma unroll
  for (int t = 0; t < 16; ++t)
#pragma unroll
    for (int r = 0; r < 8; ++r) acc[t][r] = 0.0f;

  for (int kc = 0; kc < 8; ++kc) {
    const int k = kc * 32;
    // A fragment from W: lane row = d0+lo, k-runs [k+hi*8, +8) and [k+16+hi*8, +8)
    const float* wr = W + (size_t)(d0 + lo) * C_DIM + k + hi * 8;
    float f0[8], f1[8];
    *(float4*)(f0)     = *(const float4*)(wr);
    *(float4*)(f0 + 4) = *(const float4*)(wr + 4);
    *(float4*)(f1)     = *(const float4*)(wr + 16);
    *(float4*)(f1 + 4) = *(const float4*)(wr + 20);
    v16bf af;
#pragma unroll
    for (int e = 0; e < 8; ++e) { af[e] = (bf16_t)f0[e]; af[8 + e] = (bf16_t)f1[e]; }

    // B fragments from X: lane = column (n0+t*16+lo), k-runs [k+hi*8, +8) and [k+16+hi*8, +8)
    const float* xr = X + (size_t)(k + hi * 8) * N_TOK + n0 + lo;
#pragma unroll
    for (int t = 0; t < 16; ++t) {
      v16bf bfrag;
#pragma unroll
      for (int e = 0; e < 8; ++e) {
        bfrag[e]     = (bf16_t)xr[(size_t)e * N_TOK + t * 16];
        bfrag[8 + e] = (bf16_t)xr[(size_t)(16 + e) * N_TOK + t * 16];
      }
      acc[t] = wmma_bf16(af, bfrag, acc[t]);
    }
  }

  if (rowmajor) {
    // Y[n][d]: per lane one 16B store per tile (8 consecutive d in VGPRs 0..7)
#pragma unroll
    for (int t = 0; t < 16; ++t) {
      v8bf h;
#pragma unroll
      for (int r = 0; r < 8; ++r) h[r] = (bf16_t)acc[t][r];
      *(v8bf*)(Y + (size_t)(n0 + t * 16 + lo) * C_DIM + d0 + 8 * hi) = h;
    }
  } else {
    // Y[d][n]: 16-lane-contiguous 2B stores per d-row
#pragma unroll
    for (int t = 0; t < 16; ++t)
#pragma unroll
      for (int r = 0; r < 8; ++r)
        Y[(size_t)(d0 + r + 8 * hi) * N_TOK + n0 + t * 16 + lo] = (bf16_t)acc[t][r];
  }
}

// ---------------------------------------------------------------------------
// Kernel 2: fused flash attention, one softmax branch per blockIdx.z.
// grid (36 qtiles, 8 batches, 2 branches), 128 threads (4 waves).
// Wave owns 16 queries x 256 dims; key loop in blocks of 32.
// branch 0: out = qry + w1 * softmax(Q K1^T) V       (writes d_out)
// branch 1: partial = w2 * softmax(Q K2^T) V         (writes workspace)
// ---------------------------------------------------------------------------
__global__ __launch_bounds__(128) void flash_kernel(
    const bf16_t* __restrict__ wsb, const float* __restrict__ qry,
    const float* __restrict__ mu_p, float* __restrict__ out,
    float* __restrict__ partial)
{
  const int qt = blockIdx.x, batch = blockIdx.y, br = blockIdx.z;
  const bf16_t* Q = wsb + (size_t)batch * SZ1;                               // [n][d]
  const bf16_t* K = wsb + (size_t)(1 + br) * TSZ + (size_t)batch * SZ1;      // [d][n]
  const bf16_t* V = wsb + (size_t)3 * TSZ + (size_t)batch * SZ1;             // [n][d]

  const float mu  = mu_p[0];
  const float wbr = (br == 0) ? (0.5f / (0.5f + mu)) : (mu / (0.5f + mu));

  __shared__ bf16_t Ks[256 * 40];      // K tile [d][k], pitch 40 (pad vs conflicts)
  __shared__ bf16_t Vs[32 * 264];      // V tile [k][d], pitch 264
  __shared__ bf16_t Ps[4 * 16 * 36];   // per-wave P tile [m][k], pitch 36

  const int lane = threadIdx.x & 31;
  const int wv   = threadIdx.x >> 5;
  const int lo   = lane & 15;
  const int hi   = lane >> 4;
  const int q0   = qt * 64 + wv * 16;

  // Resident Q A-fragments: 8 chunks of K=32 (contiguous 16B runs per lane)
  v16bf qf[8];
#pragma unroll
  for (int dc = 0; dc < 8; ++dc) {
    const bf16_t* qr = Q + (size_t)(q0 + lo) * C_DIM + dc * 32 + hi * 8;
    v8bf a = *(const v8bf*)(qr);
    v8bf b = *(const v8bf*)(qr + 16);
#pragma unroll
    for (int e = 0; e < 8; ++e) { qf[dc][e] = a[e]; qf[dc][8 + e] = b[e]; }
  }

  float mi[8], li[8];
#pragma unroll
  for (int r = 0; r < 8; ++r) { mi[r] = -1e30f; li[r] = 0.0f; }
  v8f o[16];
#pragma unroll
  for (int t = 0; t < 16; ++t)
#pragma unroll
    for (int r = 0; r < 8; ++r) o[t][r] = 0.0f;

  for (int kb = 0; kb < N_TOK; kb += 32) {
    __syncthreads();
    // stage K tile: 256 rows x 32 keys (16B chunks), coalesced
    for (int cidx = threadIdx.x; cidx < 1024; cidx += 128) {
      int row = cidx >> 2, ch = cidx & 3;
      *(v8bf*)(&Ks[row * 40 + ch * 8]) =
          *(const v8bf*)(K + (size_t)row * N_TOK + kb + ch * 8);
    }
    // stage V tile: 32 key-rows x 256 dims (16B chunks), coalesced, no transpose
    for (int cidx = threadIdx.x; cidx < 1024; cidx += 128) {
      int row = cidx >> 5, ch = cidx & 31;
      *(v8bf*)(&Vs[row * 264 + ch * 8]) =
          *(const v8bf*)(V + (size_t)(kb + row) * C_DIM + ch * 8);
    }
    __syncthreads();

    // S = Q K^T : two 16x16 score tiles (keys kb..kb+15, kb+16..kb+31)
    v8f s[2];
#pragma unroll
    for (int kt = 0; kt < 2; ++kt) {
      v8f st;
#pragma unroll
      for (int r = 0; r < 8; ++r) st[r] = 0.0f;
#pragma unroll
      for (int dc = 0; dc < 8; ++dc) {
        const bf16_t* kr = &Ks[(dc * 32 + hi * 8) * 40 + kt * 16 + lo];
        v16bf bf;
#pragma unroll
        for (int e = 0; e < 8; ++e) { bf[e] = kr[e * 40]; bf[8 + e] = kr[(16 + e) * 40]; }
        st = wmma_bf16(qf[dc], bf, st);
      }
      s[kt] = st;
    }

    // Online softmax per query row (each row lives in one VGPR slot across a
    // 16-lane half; xor-shuffles 1/2/4/8 stay inside the half).
    float sc[8];
#pragma unroll
    for (int r = 0; r < 8; ++r) {
      float v = fmaxf(s[0][r], s[1][r]);
      v = fmaxf(v, __shfl_xor(v, 1, 32));
      v = fmaxf(v, __shfl_xor(v, 2, 32));
      v = fmaxf(v, __shfl_xor(v, 4, 32));
      v = fmaxf(v, __shfl_xor(v, 8, 32));
      float mnew = fmaxf(mi[r], v);
      sc[r] = __expf(mi[r] - mnew);
      float p0 = __expf(s[0][r] - mnew);
      float p1 = __expf(s[1][r] - mnew);
      float ps = p0 + p1;
      ps += __shfl_xor(ps, 1, 32);
      ps += __shfl_xor(ps, 2, 32);
      ps += __shfl_xor(ps, 4, 32);
      ps += __shfl_xor(ps, 8, 32);
      li[r] = li[r] * sc[r] + ps;
      mi[r] = mnew;
      // C-layout -> [m][k] staging (wave-private region of Ps)
      bf16_t* pr = &Ps[wv * 576 + (r + 8 * hi) * 36];
      pr[lo]      = (bf16_t)p0;
      pr[16 + lo] = (bf16_t)p1;
    }
    asm volatile("s_wait_dscnt 0" ::: "memory");  // wave-local LDS RAW fence

    // rescale running accumulators
#pragma unroll
    for (int t = 0; t < 16; ++t)
#pragma unroll
      for (int r = 0; r < 8; ++r) o[t][r] *= sc[r];

    // P as A-fragment (16x32), gathered from Ps
    const bf16_t* pq = &Ps[wv * 576 + lo * 36 + hi * 8];
    v4bf p0a = *(const v4bf*)(pq);
    v4bf p0b = *(const v4bf*)(pq + 4);
    v4bf p1a = *(const v4bf*)(pq + 16);
    v4bf p1b = *(const v4bf*)(pq + 20);
    v16bf pf;
#pragma unroll
    for (int e = 0; e < 4; ++e) {
      pf[e] = p0a[e]; pf[4 + e] = p0b[e]; pf[8 + e] = p1a[e]; pf[12 + e] = p1b[e];
    }

    // O += P * V  (16 d-tiles)
#pragma unroll
    for (int t = 0; t < 16; ++t) {
      const bf16_t* vr = &Vs[(hi * 8) * 264 + t * 16 + lo];
      v16bf vf;
#pragma unroll
      for (int e = 0; e < 8; ++e) { vf[e] = vr[e * 264]; vf[8 + e] = vr[(16 + e) * 264]; }
      o[t] = wmma_bf16(pf, vf, o[t]);
    }
  }

  // Epilogue: normalize, blend weight, write transposed ([d][q] == output layout)
  float inv[8];
#pragma unroll
  for (int r = 0; r < 8; ++r) inv[r] = wbr / li[r];

#pragma unroll
  for (int t = 0; t < 16; ++t) {
    const int d = t * 16 + lo;
    const size_t base = (size_t)batch * SZ1 + (size_t)d * N_TOK + q0 + 8 * hi;
    float vals[8];
#pragma unroll
    for (int r = 0; r < 8; ++r) vals[r] = o[t][r] * inv[r];
    if (br == 0) {
      float4 qa = *(const float4*)(qry + base);
      float4 qb = *(const float4*)(qry + base + 4);
      float4 oa, ob;
      oa.x = qa.x + vals[0]; oa.y = qa.y + vals[1];
      oa.z = qa.z + vals[2]; oa.w = qa.w + vals[3];
      ob.x = qb.x + vals[4]; ob.y = qb.y + vals[5];
      ob.z = qb.z + vals[6]; ob.w = qb.w + vals[7];
      *(float4*)(out + base)     = oa;
      *(float4*)(out + base + 4) = ob;
    } else {
      float4 oa, ob;
      oa.x = vals[0]; oa.y = vals[1]; oa.z = vals[2]; oa.w = vals[3];
      ob.x = vals[4]; ob.y = vals[5]; ob.z = vals[6]; ob.w = vals[7];
      *(float4*)(partial + base)     = oa;
      *(float4*)(partial + base + 4) = ob;
    }
  }
}

// ---------------------------------------------------------------------------
// Kernel 3: out += partial (branch-1 contribution). 4718592 floats exactly.
// ---------------------------------------------------------------------------
__global__ __launch_bounds__(256) void add_kernel(float* __restrict__ out,
                                                  const float* __restrict__ partial)
{
  const size_t i = ((size_t)blockIdx.x * 256 + threadIdx.x) * 4;
  float4 a = *(const float4*)(out + i);
  const float4 b = *(const float4*)(partial + i);
  a.x += b.x; a.y += b.y; a.z += b.z; a.w += b.w;
  *(float4*)(out + i) = a;
}

extern "C" void kernel_launch(void* const* d_in, const int* in_sizes, int n_in,
                              void* d_out, int out_size, void* d_ws, size_t ws_size,
                              hipStream_t stream) {
  (void)in_sizes; (void)n_in; (void)out_size; (void)ws_size;
  const float* qry  = (const float*)d_in[0];
  const float* mask = (const float*)d_in[1];
  const float* fts  = (const float*)d_in[2];
  const float* Wq   = (const float*)d_in[3];
  const float* Wk   = (const float*)d_in[4];
  const float* Wv   = (const float*)d_in[5];
  const float* mu   = (const float*)d_in[6];
  float* out = (float*)d_out;

  bf16_t* wsb = (bf16_t*)d_ws;
  // bf16 regions: Q | K1 | K2 | V  (4 * TSZ elems * 2B = 37,748,736 bytes)
  float* partial = (float*)((char*)d_ws + (size_t)4 * TSZ * sizeof(bf16_t));

  proj_kernel<<<dim3(9, 4, 32), 128, 0, stream>>>(qry, mask, fts, Wq, Wk, Wv, wsb);
  flash_kernel<<<dim3(36, 8, 2), 128, 0, stream>>>(wsb, qry, mu, out, partial);
  add_kernel<<<dim3(4608), 256, 0, stream>>>(out, partial);
}
